// QNet_75153337745796
// MI455X (gfx1250) — matrix operand order repaired and checked by reading, not compiled, not run on hardware
//
#include <hip/hip_runtime.h>

typedef __attribute__((ext_vector_type(16))) _Float16 v16h;
typedef __attribute__((ext_vector_type(8)))  float    v8f;

#define LATENT 64
#define KDIM   128
#define HDIM   128

__global__ __launch_bounds__(256) void qnet_fused(
    const float* __restrict__ embed,
    const float* __restrict__ graph_embed,
    const int*   __restrict__ rep_idx,
    const float* __restrict__ w1,
    const float* __restrict__ b1,
    const float* __restrict__ w2,
    const float* __restrict__ b2,
    float*       __restrict__ out,
    int nTiles)
{
    // f16 copy of w1 (row-major [h][k]) staged in LDS: 32 KB of 320 KB/WGP.
    // (The compiler then hoists all 32 B fragments into VGPRs for the tile loop.)
    __shared__ alignas(32) _Float16 w1h[HDIM * KDIM];
    __shared__ float b1s[HDIM];
    __shared__ float w2s[HDIM];

    for (int i = threadIdx.x; i < HDIM * KDIM; i += 256)
        w1h[i] = (_Float16)w1[i];
    if (threadIdx.x < HDIM) {
        b1s[threadIdx.x] = b1[threadIdx.x];
        w2s[threadIdx.x] = w2[threadIdx.x];
    }
    __syncthreads();

    const float b2v = b2[0];
    const int lane = threadIdx.x & 31;
    const int col  = lane & 15;          // N-column / M-row index within tile
    const bool hi  = lane >= 16;
    const int aK   = hi ? 16 : 0;        // A fragment: K base within a 32-slice
    const int bK   = hi ? 16 : 0;        // B fragment: K base within a 32-slice
    const int pfK  = hi ? 32 : 0;        // prefetch: hi half covers row's 2nd 128B line

    const int gwave  = blockIdx.x * 8 + (threadIdx.x >> 5);
    const int nWaves = gridDim.x * 8;

    for (int tile = gwave; tile < nTiles; tile += nWaves) {
        const int m0   = tile << 4;
        const int row  = m0 + col;       // lanes L and L+16 share a row (A layout)
        const int grow = rep_idx[row];
        const float* srcA = embed       + (size_t)row  * LATENT;
        const float* srcB = graph_embed + (size_t)grow * LATENT;

        // Deep streaming prefetch of upcoming embed tiles (full 256B rows:
        // lanes 0-15 cover bytes [0,128), lanes 16-31 cover [128,256)).
        {
            const int t1 = tile + nWaves;
            const int t2 = tile + 2 * nWaves;
            if (t1 < nTiles) {
                __builtin_prefetch(embed + ((size_t)t1 * 16 + col) * LATENT + pfK, 0, 0);
                __builtin_prefetch(rep_idx + (size_t)t1 * 16, 0, 0);
            }
            if (t2 < nTiles)
                __builtin_prefetch(embed + ((size_t)t2 * 16 + col) * LATENT + pfK, 0, 0);
        }

        v8f acc[8];
        #pragma unroll
        for (int i = 0; i < 8; ++i) acc[i] = (v8f)0.0f;

        #pragma unroll
        for (int kk = 0; kk < 4; ++kk) {           // 4 K-slices of 32
            const float* src = (kk < 2) ? srcA : srcB;
            const int kbase = (kk & 1) * 32 + aK;
            // One 16-float run per lane: K = kbase..+15 (contiguous, matches B)
            float4 f0 = *(const float4*)(src + kbase);
            float4 f1 = *(const float4*)(src + kbase + 4);
            float4 f2 = *(const float4*)(src + kbase + 8);
            float4 f3 = *(const float4*)(src + kbase + 12);
            v16h a;
            a[0]=(_Float16)f0.x;  a[1]=(_Float16)f0.y;  a[2]=(_Float16)f0.z;  a[3]=(_Float16)f0.w;
            a[4]=(_Float16)f1.x;  a[5]=(_Float16)f1.y;  a[6]=(_Float16)f1.z;  a[7]=(_Float16)f1.w;
            a[8]=(_Float16)f2.x;  a[9]=(_Float16)f2.y;  a[10]=(_Float16)f2.z; a[11]=(_Float16)f2.w;
            a[12]=(_Float16)f3.x; a[13]=(_Float16)f3.y; a[14]=(_Float16)f3.z; a[15]=(_Float16)f3.w;

            #pragma unroll
            for (int ht = 0; ht < 8; ++ht) {       // 8 output h-tiles of 16
                // B fragment: lane holds w1[h = ht*16+col][K0 .. K0+15] (32B, reg-hoisted)
                const v16h bfrag = *(const v16h*)&w1h[(ht * 16 + col) * KDIM + kk * 32 + bK];
                acc[ht] = __builtin_amdgcn_wmma_f32_16x16x32_f16(
                    false, a, false, bfrag, (short)0, acc[ht], false, false);
            }
        }

        // Fused epilogue: bias + ReLU + w2 dot, reduced over the 16 N-columns.
        float s[8];
        #pragma unroll
        for (int r = 0; r < 8; ++r) s[r] = 0.0f;
        #pragma unroll
        for (int ht = 0; ht < 8; ++ht) {
            const float bb = b1s[ht * 16 + col];
            const float ww = w2s[ht * 16 + col];
            #pragma unroll
            for (int r = 0; r < 8; ++r) {
                float hv = acc[ht][r] + bb;        // C layout: M = r (+8 for hi half)
                hv = hv > 0.0f ? hv : 0.0f;
                s[r] = __builtin_fmaf(ww, hv, s[r]);
            }
        }
        // Butterfly within each 16-lane half (masks < 16 never cross halves).
        #pragma unroll
        for (int m = 1; m <= 8; m <<= 1) {
            #pragma unroll
            for (int r = 0; r < 8; ++r)
                s[r] += __shfl_xor(s[r], m, 32);
        }
        if (col == 0) {
            float* o = out + m0 + (hi ? 8 : 0);
            float4 o0 = make_float4(s[0]+b2v, s[1]+b2v, s[2]+b2v, s[3]+b2v);
            float4 o1 = make_float4(s[4]+b2v, s[5]+b2v, s[6]+b2v, s[7]+b2v);
            *(float4*)o       = o0;
            *(float4*)(o + 4) = o1;
        }
    }
}

extern "C" void kernel_launch(void* const* d_in, const int* in_sizes, int n_in,
                              void* d_out, int out_size, void* d_ws, size_t ws_size,
                              hipStream_t stream) {
    const float* embed = (const float*)d_in[0];
    const float* gemb  = (const float*)d_in[1];
    const int*   ridx  = (const int*)d_in[2];
    const float* w1    = (const float*)d_in[3];
    const float* b1    = (const float*)d_in[4];
    const float* w2    = (const float*)d_in[5];
    const float* b2    = (const float*)d_in[6];
    float* out = (float*)d_out;

    const int N      = in_sizes[0] / LATENT;   // 2,097,152
    const int nTiles = N / 16;                 // 131,072 wave-tiles
    int blocks = 2048;                         // grid-stride; amortizes w1 staging
    if (blocks * 8 > nTiles) blocks = (nTiles + 7) / 8;
    if (blocks < 1) blocks = 1;
    qnet_fused<<<blocks, 256, 0, stream>>>(embed, gemb, ridx, w1, b1, w2, b2, out, nTiles);
}
